// Attention_50294067036729
// MI455X (gfx1250) — compile-verified, matrix operand order checked by reading, not run on hardware
//
#include <hip/hip_runtime.h>
#include <hip/hip_bf16.h>

typedef __attribute__((ext_vector_type(16))) __bf16 v16bf;
typedef __attribute__((ext_vector_type(8)))  float  v8f;

#define DIMD 512
#define TT   64
#define BB   8
#define SS   512

// ---------------------------------------------------------------------------
// Hardware tanh: gfx1250 has V_TANH_F32 (TRANS op). Prefer the clang builtin
// (compiler handles TRANS hazards); fall back to inline asm with trailing
// v_nop per ISA §7.4 (1 independent op after a TRANS op before use).
// ---------------------------------------------------------------------------
__device__ __forceinline__ float fast_tanh(float x) {
#if __has_builtin(__builtin_amdgcn_tanhf)
  return __builtin_amdgcn_tanhf(x);
#elif __has_builtin(__builtin_amdgcn_tanh_f32)
  return __builtin_amdgcn_tanh_f32(x);
#else
  float y;
  asm("v_tanh_f32 %0, %1\n\tv_nop\n\tv_nop" : "=v"(y) : "v"(x));
  return y;
#endif
}

// Load a 16x32 (MxK) bf16 WMMA operand tile from a row-major f32 matrix.
// ISA 16-bit A layout (wave32): lane L -> row = L%16, g = L/16;
//   vector elems 0..7  hold K = g*8 + 0..7
//   vector elems 8..15 hold K = 16 + g*8 + 0..7
__device__ __forceinline__ v16bf load_tile_bf16(const float* __restrict__ base,
                                                int ld, int mn0, int k0, int lane) {
  const int mn = lane & 15;
  const int g  = lane >> 4;
  const float* p = base + (size_t)(mn0 + mn) * ld + k0 + g * 8;
  v16bf r;
#pragma unroll
  for (int i = 0; i < 8; ++i) r[i] = (__bf16)p[i];
#pragma unroll
  for (int i = 0; i < 8; ++i) r[8 + i] = (__bf16)p[16 + i];
  return r;
}

__device__ __forceinline__ v8f wmma_bf16(v16bf a, v16bf b, v8f c) {
  // (neg_a, A, neg_b, B, c_mod, C, reuse_a, reuse_b)
  return __builtin_amdgcn_wmma_f32_16x16x32_bf16(false, a, false, b, (short)0, c,
                                                 false, false);
}

// ---------------------------------------------------------------------------
// Kernel 1: wq = input(rows t*B+b) * Wq^T          [512 x 512]
//           uh' = mem(rows s*B+b) * Wc^T + cov_vec[b,s]*Wcov + bcov  [4096x512]
// One 16x16 output tile per wave, 8 waves per block.
// ---------------------------------------------------------------------------
__global__ void __launch_bounds__(256)
k_pre(const float* __restrict__ input, const float* __restrict__ mem,
      const float* __restrict__ cov,   const float* __restrict__ Wq,
      const float* __restrict__ Wc,    const float* __restrict__ Wcov,
      const float* __restrict__ bcov,  float* __restrict__ wq_out,
      float* __restrict__ uh_out) {
  const int lane = threadIdx.x & 31;
  const int wave = threadIdx.x >> 5;
  const int tile = blockIdx.x * 8 + wave;

  const bool is_uh = (tile >= 1024);
  const int  t2 = is_uh ? (tile - 1024) : tile;
  const int  r0 = (t2 >> 5) * 16;
  const int  e0 = (t2 & 31) * 16;
  const float* A = is_uh ? mem : input;
  const float* W = is_uh ? Wc  : Wq;
  float* out     = is_uh ? uh_out : wq_out;

  v8f acc = {};
#pragma unroll 4
  for (int k0 = 0; k0 < DIMD; k0 += 32) {
    v16bf av = load_tile_bf16(A, DIMD, r0, k0, lane);
    v16bf bv = load_tile_bf16(W, DIMD, e0, k0, lane);
    acc = wmma_bf16(av, bv, acc);
  }

  const int n = lane & 15, g = lane >> 4;
  const int e = e0 + n;
  const float wce = is_uh ? Wcov[e] : 0.0f;
  const float bce = is_uh ? bcov[e] : 0.0f;
#pragma unroll
  for (int j = 0; j < 8; ++j) {
    const int r = r0 + j + 8 * g;      // C/D layout: VGPR j, lanes>=16 -> M=8+j
    float val = acc[j];
    if (is_uh) {
      const int s = r >> 3;            // r = s*B + b, B = 8
      const int b = r & 7;
      val += cov[b * SS + s] * wce + bce;
    }
    out[(size_t)r * DIMD + e] = val;
  }
}

// ---------------------------------------------------------------------------
// Kernel 2: per (t,b): align[s] = sum_d v[d]*tanh(wq[d] + uh'[s,b,d]);
// softmax over s; write align & cov_new; c[d] = sum_s a[s]*mem[s,b,d].
// One block (8 waves / 256 threads) per row r = t*B + b.
// Lane l owns contiguous d-chunk [l*16, l*16+16) -> 4 x b128 loads per s.
// ---------------------------------------------------------------------------
__global__ void __launch_bounds__(256)
k_attn(const float* __restrict__ wq, const float* __restrict__ uh,
       const float* __restrict__ mem, const float* __restrict__ cov,
       const float* __restrict__ vvec, float* __restrict__ c_out,
       float* __restrict__ align_out, float* __restrict__ cov_out) {
  const int r    = blockIdx.x;         // t*B + b
  const int b    = r & (BB - 1);
  const int tid  = threadIdx.x;
  const int lane = tid & 31;
  const int wave = tid >> 5;

  __shared__ float a_lds[SS];
  __shared__ float sred[256];

  // Per-lane contiguous chunks of wq row and v (64B-aligned float4 loads).
  float wqr[16], vr[16];
  {
    const float4* wq4 = (const float4*)(wq + (size_t)r * DIMD + lane * 16);
    const float4* v4  = (const float4*)(vvec + lane * 16);
#pragma unroll
    for (int j = 0; j < 4; ++j) {
      float4 w = wq4[j], vv = v4[j];
      wqr[4*j+0] = w.x;  wqr[4*j+1] = w.y;  wqr[4*j+2] = w.z;  wqr[4*j+3] = w.w;
      vr [4*j+0] = vv.x; vr [4*j+1] = vv.y; vr [4*j+2] = vv.z; vr [4*j+3] = vv.w;
    }
  }

  for (int s = wave; s < SS; s += 8) {
    const float4* uh4 =
        (const float4*)(uh + (size_t)(s * BB + b) * DIMD + lane * 16);
    float acc = 0.0f;
#pragma unroll
    for (int j = 0; j < 4; ++j) {
      float4 u = uh4[j];
      acc += vr[4*j+0] * fast_tanh(wqr[4*j+0] + u.x);
      acc += vr[4*j+1] * fast_tanh(wqr[4*j+1] + u.y);
      acc += vr[4*j+2] * fast_tanh(wqr[4*j+2] + u.z);
      acc += vr[4*j+3] * fast_tanh(wqr[4*j+3] + u.w);
    }
#pragma unroll
    for (int m = 16; m >= 1; m >>= 1)
      acc += __shfl_xor(acc, m, 32);   // wave32 reduction
    if (lane == 0) a_lds[s] = acc;
  }
  __syncthreads();

  // Softmax over 512 values; each thread owns s = tid and s = tid+256.
  const float x0 = a_lds[tid], x1 = a_lds[tid + 256];
  sred[tid] = fmaxf(x0, x1);
  __syncthreads();
  for (int off = 128; off >= 1; off >>= 1) {
    if (tid < off) sred[tid] = fmaxf(sred[tid], sred[tid + off]);
    __syncthreads();
  }
  const float mx = sred[0];
  __syncthreads();
  const float e0 = __expf(x0 - mx), e1 = __expf(x1 - mx);
  sred[tid] = e0 + e1;
  __syncthreads();
  for (int off = 128; off >= 1; off >>= 1) {
    if (tid < off) sred[tid] += sred[tid + off];
    __syncthreads();
  }
  const float inv = 1.0f / sred[0];
  __syncthreads();

  const float a0 = e0 * inv, a1 = e1 * inv;
  a_lds[tid]       = a0;
  a_lds[tid + 256] = a1;
  align_out[(size_t)r * SS + tid]       = a0;
  align_out[(size_t)r * SS + tid + 256] = a1;
  cov_out[(size_t)r * SS + tid]       = cov[b * SS + tid]       + a0;
  cov_out[(size_t)r * SS + tid + 256] = cov[b * SS + tid + 256] + a1;
  __syncthreads();

  // Context: c[d] = sum_s a[s] * mem[s,b,d]; thread owns d = 2*tid, 2*tid+1.
  const int d2 = tid * 2;
  float c0 = 0.0f, c1 = 0.0f;
  for (int s = 0; s < SS; ++s) {
    const float a = a_lds[s];
    const float2 m2 =
        *(const float2*)(mem + (size_t)(s * BB + b) * DIMD + d2);
    c0 = fmaf(a, m2.x, c0);
    c1 = fmaf(a, m2.y, c1);
  }
  *(float2*)(c_out + (size_t)r * DIMD + d2) = make_float2(c0, c1);
}

// ---------------------------------------------------------------------------
// Kernel 3: attn_h = c*WoutC^T + q*WoutQ^T + bout.
// Wout is [D, 2D] row-major: columns 0..511 multiply c, 512..1023 multiply q.
// ---------------------------------------------------------------------------
__global__ void __launch_bounds__(256)
k_out(const float* __restrict__ c_mat, const float* __restrict__ input,
      const float* __restrict__ Wout, const float* __restrict__ bout,
      float* __restrict__ out) {
  const int lane = threadIdx.x & 31;
  const int wave = threadIdx.x >> 5;
  const int tile = blockIdx.x * 8 + wave;
  const int r0 = (tile >> 5) * 16;
  const int e0 = (tile & 31) * 16;

  v8f acc = {};
#pragma unroll 4
  for (int k0 = 0; k0 < DIMD; k0 += 32) {
    v16bf av = load_tile_bf16(c_mat, DIMD, r0, k0, lane);
    v16bf bv = load_tile_bf16(Wout, 2 * DIMD, e0, k0, lane);   // Wout[:, 0:512]
    acc = wmma_bf16(av, bv, acc);
  }
#pragma unroll 4
  for (int k0 = 0; k0 < DIMD; k0 += 32) {
    v16bf av = load_tile_bf16(input, DIMD, r0, k0, lane);
    v16bf bv = load_tile_bf16(Wout + DIMD, 2 * DIMD, e0, k0, lane); // Wout[:,512:]
    acc = wmma_bf16(av, bv, acc);
  }

  const int n = lane & 15, g = lane >> 4;
  const int e = e0 + n;
  const float bb = bout[e];
#pragma unroll
  for (int j = 0; j < 8; ++j) {
    const int r = r0 + j + 8 * g;
    out[(size_t)r * DIMD + e] = acc[j] + bb;   // [T,B,D] row r = t*B+b
  }
}

extern "C" void kernel_launch(void* const* d_in, const int* in_sizes, int n_in,
                              void* d_out, int out_size, void* d_ws, size_t ws_size,
                              hipStream_t stream) {
  const float* input = (const float*)d_in[0];   // [T,B,D]
  const float* mem   = (const float*)d_in[1];   // [S,B,D]
  const float* cov   = (const float*)d_in[2];   // [B,S]
  const float* Wq    = (const float*)d_in[3];   // [D,D]
  const float* Wc    = (const float*)d_in[4];   // [D,D]
  const float* Wcov  = (const float*)d_in[5];   // [D,1]
  const float* bcov  = (const float*)d_in[6];   // [D]
  const float* v     = (const float*)d_in[7];   // [1,D]
  const float* Wout  = (const float*)d_in[8];   // [D,2D]
  const float* bout  = (const float*)d_in[9];   // [D]

  float* ws    = (float*)d_ws;
  float* wq_ws = ws;                                   // 512*512   = 1 MB
  float* uh_ws = ws + (size_t)512 * 512;               // 4096*512  = 8 MB
  float* c_ws  = uh_ws + (size_t)4096 * 512;           // 512*512   = 1 MB

  float* attn_h = (float*)d_out;                       // [T,B,D]
  float* align  = attn_h + (size_t)TT * BB * DIMD;     // [T,B,S]
  float* covn   = align + (size_t)TT * BB * SS;        // [T,B,S]

  k_pre <<<1152, 256, 0, stream>>>(input, mem, cov, Wq, Wc, Wcov, bcov,
                                   wq_ws, uh_ws);
  k_attn<<<512,  256, 0, stream>>>(wq_ws, uh_ws, mem, cov, v, c_ws, align, covn);
  k_out <<<128,  256, 0, stream>>>(c_ws, input, Wout, bout, attn_h);
}
